// selfAttention_41291815583764
// MI455X (gfx1250) — matrix operand order repaired and checked by reading, not compiled, not run on hardware
//
#include <hip/hip_runtime.h>
#include <hip/hip_bf16.h>

// ---------------- problem constants ----------------
enum { Bn = 4, Ln = 1024, En = 1024, Hn = 16, Dn = 64 };
// softmax scale = 1/sqrt(E) = 1/32
#define SCALE 0.03125f
#define MASKED_VAL (-3.125e18f)   // -1e20 * (1/32)

// ---------------- WMMA types & helpers ----------------
typedef __attribute__((ext_vector_type(16))) __bf16 v16bf;
typedef __attribute__((ext_vector_type(8)))  float  v8f;

union FragU {
    v16bf v;
    __bf16 h[16];
    uint4 q[2];
};

// A-fragment (16x32 bf16, M x K), rowbase = pointer to this lane's row (r0 + lane%16).
// lanes 0-15: K = kc+{0..7, 16..23}; lanes 16-31: K = kc+{8..15, 24..31}.
static __device__ __forceinline__ v16bf loadA_bf(const __bf16* rowbase, int kc, int half) {
    FragU f;
    f.q[0] = *reinterpret_cast<const uint4*>(rowbase + kc + half * 8);
    f.q[1] = *reinterpret_cast<const uint4*>(rowbase + kc + 16 + half * 8);
    return f.v;
}

// Same A layout but source is fp32 (convert on the fly, native v_cvt_pk_bf16_f32).
static __device__ __forceinline__ v16bf loadA_f32(const float* rowbase, int kc, int half) {
    FragU f;
    const float* p0 = rowbase + kc + half * 8;
    const float* p1 = rowbase + kc + 16 + half * 8;
#pragma unroll
    for (int i = 0; i < 8; ++i) f.h[i]     = (__bf16)p0[i];
#pragma unroll
    for (int i = 0; i < 8; ++i) f.h[8 + i] = (__bf16)p1[i];
    return f.v;
}

// B-fragment (32x16 bf16, K x N). Source stores logical column n as a contiguous
// K-vector (row n of the source array); colbase = pointer to column n's vector.
// lanes 0-15: K = kc+{0..7, 16..23}; lanes 16-31: K = kc+{8..15, 24..31}.
static __device__ __forceinline__ v16bf loadB_bf(const __bf16* colbase, int kc, int half) {
    FragU f;
    f.q[0] = *reinterpret_cast<const uint4*>(colbase + kc + half * 8);
    f.q[1] = *reinterpret_cast<const uint4*>(colbase + kc + 16 + half * 8);
    return f.v;
}

static __device__ __forceinline__ v8f wmma_bf16(v16bf a, v16bf b, v8f c) {
    return __builtin_amdgcn_wmma_f32_16x16x32_bf16(false, a, false, b, (short)0, c, false, false);
}

// ---------------- kernel 0: fp32 -> bf16 weight conversion ----------------
__global__ void cvt_bf16_kernel(const float* __restrict__ src, __bf16* __restrict__ dst, int n) {
    int i = (blockIdx.x * 256 + threadIdx.x) * 2;
    if (i < n) {
        dst[i]     = (__bf16)src[i];
        dst[i + 1] = (__bf16)src[i + 1];   // n is always even here
    }
}

// ---------------- kernel 1: per-head QKV projection ----------------
// grid = B*H*(L/16), block = 96 (3 waves: wave0->Q, wave1->K, wave2->V)
// Qp,Kp: bf16 [B,H,L,D] row-major. Vt: bf16 [B,H,D,L] (transposed for PV B-fragments).
__global__ __launch_bounds__(96) void proj_kernel(
    const float* __restrict__ vals, const float* __restrict__ keys, const float* __restrict__ qrys,
    const __bf16* __restrict__ Wv_bf, const __bf16* __restrict__ Wk_bf,
    const __bf16* __restrict__ Wq_bf,
    __bf16* __restrict__ Qp, __bf16* __restrict__ Kp, __bf16* __restrict__ Vt) {
    int idx = blockIdx.x;
    int lt = idx % (Ln / 16);
    int h  = (idx / (Ln / 16)) % Hn;
    int b  = idx / ((Ln / 16) * Hn);
    int l0 = lt * 16;

    int tid = threadIdx.x;
    int wave = tid >> 5;
    int lane = tid & 31;
    int half = lane >> 4, l16 = lane & 15;

    const float* srcs[3] = { qrys, keys, vals };
    const __bf16* wts[3] = { Wq_bf, Wk_bf, Wv_bf };
    const float* x = srcs[wave];
    const __bf16* W = wts[wave];

    // x[b, l0+l16, h*64 + k]
    const float* xrow = x + ((size_t)b * Ln + l0 + l16) * En + h * Dn;
    v16bf a0 = loadA_f32(xrow, 0,  half);
    v16bf a1 = loadA_f32(xrow, 32, half);

    size_t bh = (size_t)b * Hn + h;
#pragma unroll
    for (int nt = 0; nt < 4; ++nt) {
        // out[m, d] = sum_k x[m,k] * W[d,k]  -> B column d = W row d (contiguous)
        const __bf16* wrow = W + (size_t)(nt * 16 + l16) * Dn;
        v8f c = {};
        c = wmma_bf16(a0, loadB_bf(wrow, 0,  half), c);
        c = wmma_bf16(a1, loadB_bf(wrow, 32, half), c);
        if (wave == 2) {
            // Vt[b,h,d,l]: lane holds fixed d, 8 consecutive l -> one 16B store
            int d = nt * 16 + l16;
            union { uint4 q; __bf16 s[8]; } pk;
#pragma unroll
            for (int j = 0; j < 8; ++j) pk.s[j] = (__bf16)c[j];
            *reinterpret_cast<uint4*>(Vt + (bh * Dn + d) * Ln + l0 + half * 8) = pk.q;
        } else {
            __bf16* dst = (wave == 0) ? Qp : Kp;
#pragma unroll
            for (int j = 0; j < 8; ++j) {
                int m = j + half * 8;
                dst[(bh * Ln + l0 + m) * Dn + nt * 16 + l16] = (__bf16)c[j];
            }
        }
    }
}

// ---------------- kernel 2: attention (scores -> softmax -> PV) ----------------
// grid = B*H*(L/16), block = 128 (4 waves). LDS score tile 16 x 1024.
#define SPITCH 1036   // f32 pitch: 1036 % 64 = 12, gcd(12,64)=4 -> 16 distinct banks
#define PPITCH 1032   // u16 pitch: 2064B rows keep 16B alignment; dword stride 516 -> conflict-free
__global__ __launch_bounds__(128) void attn_kernel(
    const __bf16* __restrict__ Qp, const __bf16* __restrict__ Kp,
    const __bf16* __restrict__ Vt, const int* __restrict__ mask,
    __bf16* __restrict__ Obuf) {
    __shared__ __align__(16) float  S[16 * SPITCH];
    __shared__ __align__(16) __bf16 P[16 * PPITCH];
    __shared__ float red[16 * 8];
    __shared__ float recipS[16];

    int idx = blockIdx.x;
    int lt = idx % (Ln / 16);
    int h  = (idx / (Ln / 16)) % Hn;
    int b  = idx / ((Ln / 16) * Hn);
    int l0 = lt * 16;

    int tid = threadIdx.x;
    int wave = tid >> 5;
    int lane = tid & 31;
    int half = lane >> 4, l16 = lane & 15;
    size_t bh = (size_t)b * Hn + h;

    // Q A-fragments for this lane's query row (fixed for whole block of k)
    const __bf16* qrow = Qp + (bh * Ln + l0 + l16) * Dn;
    v16bf qa0 = loadA_bf(qrow, 0,  half);
    v16bf qa1 = loadA_bf(qrow, 32, half);

    // ---- phase 1: S = mask(Q K^T) * scale, 16-col k-tiles split across 4 waves
    for (int kt = wave * 16; kt < Ln; kt += 64) {
        const __bf16* krow = Kp + (bh * Ln + kt + l16) * Dn;  // B column k = K' row k
        // prefetch next k-tile for this wave (global_prefetch_b8)
        if (kt + 64 < Ln) __builtin_prefetch(krow + (size_t)64 * Dn, 0, 3);
        v8f c = {};
        c = wmma_bf16(qa0, loadB_bf(krow, 0,  half), c);
        c = wmma_bf16(qa1, loadB_bf(krow, 32, half), c);
        int mk = mask[(size_t)b * Ln + kt + l16];
#pragma unroll
        for (int j = 0; j < 8; ++j) {
            int m = j + half * 8;
            S[m * SPITCH + kt + l16] = mk ? c[j] * SCALE : MASKED_VAL;
        }
    }
    __syncthreads();

    // ---- phase 2: two-pass softmax; row = tid&15, 8 segments of 128 cols
    {
        int row = tid & 15, seg = tid >> 4;
        const float* srow = S + row * SPITCH + seg * 128;
        float mx = -3.4e38f;
        for (int c2 = 0; c2 < 128; ++c2) mx = fmaxf(mx, srow[c2]);
        red[row * 8 + seg] = mx;
        __syncthreads();
        float rmax = red[row * 8 + 0];
#pragma unroll
        for (int s2 = 1; s2 < 8; ++s2) rmax = fmaxf(rmax, red[row * 8 + s2]);
        __syncthreads();
        float sum = 0.f;
        __bf16* prow = P + row * PPITCH + seg * 128;
        for (int c2 = 0; c2 < 128; ++c2) {
            float p = __expf(srow[c2] - rmax);
            sum += p;
            prow[c2] = (__bf16)p;   // unnormalized; normalize at output
        }
        red[row * 8 + seg] = sum;
        __syncthreads();
        if (seg == 0) {
            float t = 0.f;
#pragma unroll
            for (int s2 = 0; s2 < 8; ++s2) t += red[row * 8 + s2];
            recipS[row] = 1.0f / t;
        }
    }
    __syncthreads();

    // ---- phase 3: O = P V', one 16-wide d-tile per wave
    {
        int nt = wave;
        const __bf16* vrow = Vt + (bh * Dn + nt * 16 + l16) * Ln; // B column d = Vt row d
        const __bf16* prow = P + l16 * PPITCH;                    // A row = query row
        v8f c = {};
#pragma unroll 4
        for (int kc = 0; kc < Ln; kc += 32) {
            if (kc + 256 < Ln) __builtin_prefetch(vrow + kc + 256, 0, 3);
            c = wmma_bf16(loadA_bf(prow, kc, half), loadB_bf(vrow, kc, half), c);
        }
        int d = h * Dn + nt * 16 + l16;   // merged-head column
#pragma unroll
        for (int j = 0; j < 8; ++j) {
            int m = j + half * 8;
            Obuf[((size_t)b * Ln + l0 + m) * En + d] = (__bf16)(c[j] * recipS[m]);
        }
    }
}

// ---------------- kernel 3: output projection  out = O @ Wo^T + bo ----------------
// grid = (B*L/64) * (E/64), block = 128. Block tile 64x64; each wave owns a 32x32
// sub-tile (2 A-frags x 2 B-frags -> 4 WMMAs per k-step => 1 fragment load per WMMA).
__global__ __launch_bounds__(128) void out_kernel(
    const __bf16* __restrict__ Obuf, const __bf16* __restrict__ Wo_bf,
    const float* __restrict__ bo, float* __restrict__ out) {
    int idx = blockIdx.x;
    int nb = idx % (En / 64);
    int mt = idx / (En / 64);
    int m0 = mt * 64, n0 = nb * 64;

    int tid = threadIdx.x;
    int wave = tid >> 5;
    int lane = tid & 31;
    int half = lane >> 4, l16 = lane & 15;

    int wr = wave >> 1, wc = wave & 1;        // 2x2 wave grid
    int mbase = m0 + wr * 32;
    int nbase = n0 + wc * 32;

    const __bf16* arow0 = Obuf  + (size_t)(mbase + l16) * En;
    const __bf16* arow1 = Obuf  + (size_t)(mbase + 16 + l16) * En;
    const __bf16* wrow0 = Wo_bf + (size_t)(nbase + l16) * En;        // B col n = Wo row n
    const __bf16* wrow1 = Wo_bf + (size_t)(nbase + 16 + l16) * En;

    v8f c00 = {}, c01 = {}, c10 = {}, c11 = {};
#pragma unroll 2
    for (int kc = 0; kc < En; kc += 32) {
        if (kc + 256 < En) {
            __builtin_prefetch(arow0 + kc + 256, 0, 3);
            __builtin_prefetch(wrow0 + kc + 256, 0, 3);
        }
        v16bf a0 = loadA_bf(arow0, kc, half);
        v16bf a1 = loadA_bf(arow1, kc, half);
        v16bf b0 = loadB_bf(wrow0, kc, half);
        v16bf b1 = loadB_bf(wrow1, kc, half);
        c00 = wmma_bf16(a0, b0, c00);
        c01 = wmma_bf16(a0, b1, c01);
        c10 = wmma_bf16(a1, b0, c10);
        c11 = wmma_bf16(a1, b1, c11);
    }
    float bias0 = bo[nbase + l16];
    float bias1 = bo[nbase + 16 + l16];
#pragma unroll
    for (int j = 0; j < 8; ++j) {
        int m = mbase + j + half * 8;
        out[(size_t)m * En + nbase + l16]             = c00[j] + bias0;
        out[(size_t)m * En + nbase + 16 + l16]        = c01[j] + bias1;
        out[(size_t)(m + 16) * En + nbase + l16]      = c10[j] + bias0;
        out[(size_t)(m + 16) * En + nbase + 16 + l16] = c11[j] + bias1;
    }
}

// ---------------- launch ----------------
extern "C" void kernel_launch(void* const* d_in, const int* in_sizes, int n_in,
                              void* d_out, int out_size, void* d_ws, size_t ws_size,
                              hipStream_t stream) {
    const float* values  = (const float*)d_in[0];
    const float* keys    = (const float*)d_in[1];
    const float* queries = (const float*)d_in[2];
    const int*   mask    = (const int*)d_in[3];
    const float* Wv      = (const float*)d_in[4];
    const float* Wk      = (const float*)d_in[5];
    const float* Wq      = (const float*)d_in[6];
    const float* Wo      = (const float*)d_in[7];
    const float* bo      = (const float*)d_in[8];
    float* out = (float*)d_out;

    // carve workspace (256B aligned slices)
    char* ws = (char*)d_ws;
    size_t off = 0;
    auto carve = [&](size_t bytes) -> char* {
        char* p = ws + off;
        off += (bytes + 255) & ~(size_t)255;
        return p;
    };
    __bf16* Wq_bf = (__bf16*)carve((size_t)Dn * Dn * 2);
    __bf16* Wk_bf = (__bf16*)carve((size_t)Dn * Dn * 2);
    __bf16* Wv_bf = (__bf16*)carve((size_t)Dn * Dn * 2);
    __bf16* Wo_bf = (__bf16*)carve((size_t)En * En * 2);
    __bf16* Qp    = (__bf16*)carve((size_t)Bn * Hn * Ln * Dn * 2);
    __bf16* Kp    = (__bf16*)carve((size_t)Bn * Hn * Ln * Dn * 2);
    __bf16* Vt    = (__bf16*)carve((size_t)Bn * Hn * Ln * Dn * 2);
    __bf16* Obuf  = (__bf16*)carve((size_t)Bn * Ln * En * 2);

    // 0) weights -> bf16 (2 elems / thread)
    {
        int nw = Dn * Dn;
        cvt_bf16_kernel<<<(nw / 2 + 255) / 256, 256, 0, stream>>>(Wq, Wq_bf, nw);
        cvt_bf16_kernel<<<(nw / 2 + 255) / 256, 256, 0, stream>>>(Wk, Wk_bf, nw);
        cvt_bf16_kernel<<<(nw / 2 + 255) / 256, 256, 0, stream>>>(Wv, Wv_bf, nw);
        int no = En * En;
        cvt_bf16_kernel<<<(no / 2 + 255) / 256, 256, 0, stream>>>(Wo, Wo_bf, no);
    }

    // 1) per-head QKV projections
    proj_kernel<<<Bn * Hn * (Ln / 16), 96, 0, stream>>>(values, keys, queries,
                                                        Wv_bf, Wk_bf, Wq_bf, Qp, Kp, Vt);

    // 2) attention
    attn_kernel<<<Bn * Hn * (Ln / 16), 128, 0, stream>>>(Qp, Kp, Vt, mask, Obuf);

    // 3) output projection + bias
    out_kernel<<<(Bn * Ln / 64) * (En / 64), 128, 0, stream>>>(Obuf, Wo_bf, bo, out);
}
